// ExpertAttention_11063835754754
// MI455X (gfx1250) — matrix operand order repaired and checked by reading, not compiled, not run on hardware
//
#include <hip/hip_runtime.h>
#include <hip/hip_bf16.h>
#include <math.h>

// ---------------------------------------------------------------------------
// ExpertAttention for MI455X (gfx1250, wave32, WMMA).
// All GEMMs use v_wmma_f32_16x16x32_f16 (f16 inputs, f32 accumulation).
// Only the routed expert is computed per batch sample.
// Round 2: software-pipelined GEMM inner loops (double-buffered fragments),
// V stored head-transposed so every WMMA operand is two b128 loads.
// ---------------------------------------------------------------------------

#define BB 128
#define SS 128
#define HH 768
#define EE 2
#define NH 12
#define HD 64

typedef __attribute__((ext_vector_type(16))) _Float16 v16h;
typedef __attribute__((ext_vector_type(8)))  float    v8f;

__device__ __forceinline__ v8f wmma_f16(v16h a, v16h b, v8f c) {
  // (neg_a, A, neg_b, B, c_mod, C, reuse_a, reuse_b)
  return __builtin_amdgcn_wmma_f32_16x16x32_f16(false, a, false, b, (short)0, c,
                                                false, false);
}

// A-matrix fragment, 16x32 f16, row-major source with leading dim lda.
// Contiguous k order (same as load_b_rows): lane = row (mod 16); lanes 0-15 hold K=0-15; lanes 16-31
// hold K=16-31 (one contiguous 16-element run per lane).
__device__ __forceinline__ v16h load_a(const _Float16* __restrict__ base,
                                       int lda, int lane) {
  const _Float16* p = base + (size_t)(lane & 15) * lda + ((lane >> 4) << 4);
  v16h a;
#pragma unroll
  for (int j = 0; j < 16; ++j) {
    a[j] = p[j];            // K = kb + 16*hi + j
  }
  return a;
}

// B-matrix fragment, 32x16 f16, when B is stored TRANSPOSED (Bt[n][k], row
// stride ldb). Contiguous k order (same as load_a): lane = column (mod 16); lanes 0-15 hold K=0-15,
// lanes 16-31 hold K=16-31 -> one contiguous 16-element run per lane.
__device__ __forceinline__ v16h load_b_rows(const _Float16* __restrict__ bt,
                                            int ldb, int lane) {
  const _Float16* p = bt + (size_t)(lane & 15) * ldb + ((lane >> 4) << 4);
  v16h v;
#pragma unroll
  for (int j = 0; j < 16; ++j) v[j] = p[j];
  return v;
}

// --------------------------- prep kernels ----------------------------------

__global__ void cvt_x_kernel(const float* __restrict__ x,
                             _Float16* __restrict__ xh, int n) {
  int i = blockIdx.x * blockDim.x + threadIdx.x;
  if (i < n) xh[i] = (_Float16)x[i];
}

// Wt layout: [mat(4)][e(2)][d(768)][h(768)] = transposed weights in f16.
__global__ void cvt_w_kernel(const float* __restrict__ Wq,
                             const float* __restrict__ Wk,
                             const float* __restrict__ Wv,
                             const float* __restrict__ Wo,
                             _Float16* __restrict__ wt) {
  size_t i = (size_t)blockIdx.x * blockDim.x + threadIdx.x;
  const size_t total = (size_t)4 * EE * HH * HH;
  if (i >= total) return;
  int h = (int)(i % HH);
  size_t r = i / HH;
  int d = (int)(r % HH); r /= HH;
  int e = (int)(r % EE);
  int m = (int)(r / EE);
  const float* src = (m == 0) ? Wq : (m == 1) ? Wk : (m == 2) ? Wv : Wo;
  wt[i] = (_Float16)src[((size_t)e * HH + h) * HH + d];
}

// Routing: mean over seq, nearest of 2 centers (L2), first-index ties.
__global__ void routing_kernel(const float* __restrict__ rs,
                               const float* __restrict__ centers,
                               int* __restrict__ assign) {
  int b = blockIdx.x;
  int t = threadIdx.x;
  float pd0 = 0.f, pd1 = 0.f;
#pragma unroll
  for (int j = 0; j < 3; ++j) {
    int h = t + 256 * j;
    float sum = 0.f;
    for (int s = 0; s < SS; ++s)
      sum += rs[((size_t)b * SS + s) * HH + h];
    float mean = sum * (1.0f / (float)SS);
    float d0 = mean - centers[h];
    float d1 = mean - centers[HH + h];
    pd0 += d0 * d0;
    pd1 += d1 * d1;
  }
  __shared__ float r0[256], r1[256];
  r0[t] = pd0; r1[t] = pd1;
  __syncthreads();
  for (int off = 128; off > 0; off >>= 1) {
    if (t < off) { r0[t] += r0[t + off]; r1[t] += r1[t + off]; }
    __syncthreads();
  }
  if (t == 0) assign[b] = (r0[0] <= r1[0]) ? 0 : 1;
}

// ----------------- software-pipelined 16x64 GEMM strip ---------------------
// acc[4] += A(16 x HH) @ B(HH x 64), A row-major (ld HH), B given transposed
// (Bt rows = output columns, ld HH). Double-buffered fragments: all loads of
// iteration k+1 are issued before the WMMAs of iteration k.
__device__ __forceinline__ void gemm_strip_768(const _Float16* __restrict__ abase,
                                               const _Float16* __restrict__ btbase,
                                               int lane, v8f acc[4]) {
  v16h a_cur = load_a(abase, HH, lane);
  v16h b_cur[4];
#pragma unroll
  for (int t = 0; t < 4; ++t)
    b_cur[t] = load_b_rows(btbase + (size_t)(16 * t) * HH, HH, lane);

  for (int k0 = 0; k0 < HH - 32; k0 += 32) {
    int kn = k0 + 32;
    v16h a_nxt = load_a(abase + kn, HH, lane);
    v16h b_nxt[4];
#pragma unroll
    for (int t = 0; t < 4; ++t)
      b_nxt[t] = load_b_rows(btbase + (size_t)(16 * t) * HH + kn, HH, lane);
#pragma unroll
    for (int t = 0; t < 4; ++t)
      acc[t] = wmma_f16(a_cur, b_cur[t], acc[t]);
    a_cur = a_nxt;
#pragma unroll
    for (int t = 0; t < 4; ++t) b_cur[t] = b_nxt[t];
  }
#pragma unroll
  for (int t = 0; t < 4; ++t)
    acc[t] = wmma_f16(a_cur, b_cur[t], acc[t]);
}

// --------------------------- QKV projection --------------------------------
// grid = B * 3 * 12 ; block = 256 (8 waves). Wave w does rows [16w,16w+16)
// of a 128x64 output tile; K loop over 768 in steps of 32.
// Q is pre-scaled by 1/sqrt(hd); V is written head-transposed:
//   Vt[b][head][hd][S]  so the ctx GEMM can use contiguous B-fragments.
__global__ void qkv_proj_kernel(const _Float16* __restrict__ Xh,
                                const _Float16* __restrict__ Wt,
                                const float* __restrict__ bq,
                                const float* __restrict__ bk,
                                const float* __restrict__ bv,
                                const int* __restrict__ assign,
                                _Float16* __restrict__ Qw,
                                _Float16* __restrict__ Kw,
                                _Float16* __restrict__ Vt) {
  int blk = blockIdx.x;
  int b   = blk / 36;
  int rem = blk % 36;
  int mat = rem / 12;
  int ct  = rem % 12;
  int wave = threadIdx.x >> 5, lane = threadIdx.x & 31;
  int hi = lane >> 4, cc = lane & 15;

  int e = assign[b];
  const _Float16* wbase = Wt + ((size_t)mat * EE + e) * HH * HH;
  const float* bias = (mat == 0) ? bq : (mat == 1) ? bk : bv;
  float scale = (mat == 0) ? 0.125f : 1.0f;  // fold 1/sqrt(hd) into Q

  int r0 = wave * 16, c0 = ct * 64;
  const _Float16* abase = Xh + ((size_t)b * SS + r0) * HH;

  v8f acc[4] = {};
  gemm_strip_768(abase, wbase + (size_t)c0 * HH, lane, acc);

#pragma unroll
  for (int t = 0; t < 4; ++t) {
    int col = c0 + 16 * t + cc;
    float bsv = bias[(size_t)e * HH + col];
    if (mat == 2) {
      // V: store transposed per head: Vt[((b*NH + head)*HD + dloc)*SS + s]
      int head = col / HD, dloc = col % HD;
      _Float16* vrow = Vt + (((size_t)b * NH + head) * HD + dloc) * SS;
#pragma unroll
      for (int i = 0; i < 8; ++i)
        vrow[r0 + i + 8 * hi] = (_Float16)(acc[t][i] + bsv);
    } else {
      _Float16* out = (mat == 0) ? Qw : Kw;
#pragma unroll
      for (int i = 0; i < 8; ++i) {
        float v = (acc[t][i] + bsv) * scale;
        out[((size_t)b * SS + r0 + i + 8 * hi) * HH + col] = (_Float16)v;
      }
    }
  }
}

// --------------------------- attention core --------------------------------
// grid = B * NH ; block = 256 (8 waves). Wave w owns score rows [16w,16w+16)
// entirely (all 128 columns) -> softmax stays inside the wave. P is staged
// through LDS to convert C-tile layout into A-fragment layout for P @ V.
__global__ void attn_kernel(const _Float16* __restrict__ Qw,
                            const _Float16* __restrict__ Kw,
                            const _Float16* __restrict__ Vt,
                            const float* __restrict__ amask,
                            _Float16* __restrict__ Cw) {
  int b = blockIdx.x / NH;
  int head = blockIdx.x % NH;
  int wave = threadIdx.x >> 5, lane = threadIdx.x & 31;
  int hi = lane >> 4, cc = lane & 15;
  int r0 = wave * 16;

  __shared__ __align__(16) _Float16 P[8][16][128];

  // scores = Qh (128x64, pre-scaled) @ Kh^T (64x128); Kh rows ARE Bt rows,
  // so B-fragments are contiguous. Batch all 8 tile fragments per k-step.
  const _Float16* qbase = Qw + ((size_t)b * SS + r0) * HH + head * HD;
  v8f acc[8] = {};
#pragma unroll
  for (int k0 = 0; k0 < HD; k0 += 32) {
    v16h a = load_a(qbase + k0, HH, lane);
    v16h bf[8];
#pragma unroll
    for (int t = 0; t < 8; ++t)
      bf[t] = load_b_rows(Kw + ((size_t)b * SS + 16 * t) * HH + head * HD + k0,
                          HH, lane);
#pragma unroll
    for (int t = 0; t < 8; ++t) acc[t] = wmma_f16(a, bf[t], acc[t]);
  }
  // BERT-style additive mask on key positions.
#pragma unroll
  for (int t = 0; t < 8; ++t) {
    float m = amask[(size_t)b * SS + 16 * t + cc];
    float madd = (1.0f - m) * -10000.0f;
#pragma unroll
    for (int i = 0; i < 8; ++i) acc[t][i] += madd;
  }
  // Row softmax. Row (i + 8*hi) lives across the 8 tiles in the 16 lanes of
  // this lane-half; xor masks 1,2,4,8 never cross the half boundary.
#pragma unroll
  for (int i = 0; i < 8; ++i) {
    float m = -3.4e38f;
#pragma unroll
    for (int t = 0; t < 8; ++t) m = fmaxf(m, acc[t][i]);
#pragma unroll
    for (int s = 1; s < 16; s <<= 1) m = fmaxf(m, __shfl_xor(m, s, 32));
    float sum = 0.f;
#pragma unroll
    for (int t = 0; t < 8; ++t) {
      float p = __expf(acc[t][i] - m);
      acc[t][i] = p;
      sum += p;
    }
#pragma unroll
    for (int s = 1; s < 16; s <<= 1) sum += __shfl_xor(sum, s, 32);
    float inv = 1.0f / sum;
#pragma unroll
    for (int t = 0; t < 8; ++t) acc[t][i] *= inv;
  }
  // C-layout -> LDS (row-major 16x128 per wave).
#pragma unroll
  for (int t = 0; t < 8; ++t)
#pragma unroll
    for (int i = 0; i < 8; ++i)
      P[wave][i + 8 * hi][16 * t + cc] = (_Float16)acc[t][i];
  __syncthreads();

  // ctx strip = P (16x128) @ Vh (128x64). Vt rows are Bt rows (ld = SS):
  // both operands are contiguous fragment loads.
  const _Float16* vtbase = Vt + ((size_t)b * NH + head) * HD * SS;
  v8f ac2[4] = {};
#pragma unroll
  for (int k0 = 0; k0 < SS; k0 += 32) {
    v16h a = load_a(&P[wave][0][k0], 128, lane);
    v16h bf[4];
#pragma unroll
    for (int u = 0; u < 4; ++u)
      bf[u] = load_b_rows(vtbase + (size_t)(16 * u) * SS + k0, SS, lane);
#pragma unroll
    for (int u = 0; u < 4; ++u) ac2[u] = wmma_f16(a, bf[u], ac2[u]);
  }
#pragma unroll
  for (int u = 0; u < 4; ++u)
#pragma unroll
    for (int i = 0; i < 8; ++i)
      Cw[((size_t)b * SS + r0 + i + 8 * hi) * HH + head * HD + 16 * u + cc] =
          (_Float16)ac2[u][i];
}

// --------------------------- output projection -----------------------------
__global__ void out_proj_kernel(const _Float16* __restrict__ Cw,
                                const _Float16* __restrict__ Wt,
                                const float* __restrict__ bo,
                                const int* __restrict__ assign,
                                float* __restrict__ out) {
  int b  = blockIdx.x / 12;
  int ct = blockIdx.x % 12;
  int wave = threadIdx.x >> 5, lane = threadIdx.x & 31;
  int hi = lane >> 4, cc = lane & 15;

  int e = assign[b];
  const _Float16* wbase = Wt + ((size_t)3 * EE + e) * HH * HH;  // mat 3 = Wo

  int r0 = wave * 16, c0 = ct * 64;
  const _Float16* abase = Cw + ((size_t)b * SS + r0) * HH;

  v8f acc[4] = {};
  gemm_strip_768(abase, wbase + (size_t)c0 * HH, lane, acc);

#pragma unroll
  for (int t = 0; t < 4; ++t) {
    int col = c0 + 16 * t + cc;
    float bsv = bo[(size_t)e * HH + col];
#pragma unroll
    for (int i = 0; i < 8; ++i)
      out[((size_t)b * SS + r0 + i + 8 * hi) * HH + col] = acc[t][i] + bsv;
  }
}

// ---------------------------------------------------------------------------

extern "C" void kernel_launch(void* const* d_in, const int* in_sizes, int n_in,
                              void* d_out, int out_size, void* d_ws,
                              size_t ws_size, hipStream_t stream) {
  const float* hidden  = (const float*)d_in[0];
  const float* amask   = (const float*)d_in[1];
  const float* routing = (const float*)d_in[2];
  const float* centers = (const float*)d_in[3];
  const float* Wq = (const float*)d_in[4];
  const float* bq = (const float*)d_in[5];
  const float* Wk = (const float*)d_in[6];
  const float* bk = (const float*)d_in[7];
  const float* Wv = (const float*)d_in[8];
  const float* bv = (const float*)d_in[9];
  const float* Wo = (const float*)d_in[10];
  const float* bo = (const float*)d_in[11];
  float* out = (float*)d_out;

  constexpr size_t NX = (size_t)BB * SS * HH;           // 12,582,912 elems
  constexpr size_t NW = (size_t)4 * EE * HH * HH;       // 4,718,592 elems

  // workspace layout (bytes); everything 16B aligned
  size_t off = 0;
  int* assign = (int*)((char*)d_ws + off);  off += 512;
  _Float16* Xh = (_Float16*)((char*)d_ws + off); off += NX * 2;
  _Float16* Wt = (_Float16*)((char*)d_ws + off); off += NW * 2;
  _Float16* Qw = (_Float16*)((char*)d_ws + off); off += NX * 2;
  _Float16* Kw = (_Float16*)((char*)d_ws + off); off += NX * 2;
  _Float16* Vt = (_Float16*)((char*)d_ws + off); off += NX * 2;
  _Float16* Cw = (_Float16*)((char*)d_ws + off); off += NX * 2;
  (void)ws_size; (void)in_sizes; (void)n_in; (void)out_size;

  cvt_x_kernel<<<(int)((NX + 255) / 256), 256, 0, stream>>>(hidden, Xh, (int)NX);
  cvt_w_kernel<<<(int)((NW + 255) / 256), 256, 0, stream>>>(Wq, Wk, Wv, Wo, Wt);
  routing_kernel<<<BB, 256, 0, stream>>>(routing, centers, assign);
  qkv_proj_kernel<<<BB * 36, 256, 0, stream>>>(Xh, Wt, bq, bk, bv, assign,
                                               Qw, Kw, Vt);
  attn_kernel<<<BB * NH, 256, 0, stream>>>(Qw, Kw, Vt, amask, Cw);
  out_proj_kernel<<<BB * 12, 256, 0, stream>>>(Cw, Wt, bo, assign, out);
}
